// GATModel_49959059587119
// MI455X (gfx1250) — compile-verified, matrix operand order checked by reading, not run on hardware
//
#include <hip/hip_runtime.h>
#include <hip/hip_bf16.h>
#include <math.h>

// ---------------------------------------------------------------------------
// GAT forward for MI455X (gfx1250).
// Dense projections run on V_WMMA_F32_16X16X32_BF16 (bf16 A/B, f32 acc).
// Edge softmax/aggregation are bandwidth-bound scalar kernels with f32 atomics.
// ---------------------------------------------------------------------------

typedef __attribute__((ext_vector_type(16))) __bf16 v16bf;
typedef __attribute__((ext_vector_type(8)))  __bf16 v8bf;
typedef __attribute__((ext_vector_type(8)))  float  v8f;

#define N_NODES   20000
#define N_EDGES   320000
#define HEADS     16
#define E_TOT     (N_EDGES + N_NODES)   // edges + self loops
#define EH_TOT    (E_TOT * HEADS)
#define NEG_SLOPE 0.2f

// ---------------------------- utility kernels ------------------------------

__global__ void k_fill_f32(float* __restrict__ p, float v, int n) {
  int i = blockIdx.x * blockDim.x + threadIdx.x;
  if (i < n) p[i] = v;
}

__global__ void k_cvt_bf16(const float* __restrict__ in, __bf16* __restrict__ out, int n) {
  int i = blockIdx.x * blockDim.x + threadIdx.x;
  if (i < n) out[i] = (__bf16)in[i];
}

// W is [K, Nout] row-major (fp32); produce Wt [Nout, K] row-major bf16 so that
// B fragments (one column of B per lane) load contiguously along K.
__global__ void k_cvt_w_t(const float* __restrict__ W, __bf16* __restrict__ Wt,
                          int K, int Nout) {
  int i = blockIdx.x * blockDim.x + threadIdx.x;
  if (i >= K * Nout) return;
  int k = i / Nout, n = i % Nout;
  Wt[(size_t)n * K + k] = (__bf16)W[i];
}

// ------------------------------- layer 1 GEMM ------------------------------
// K = 3: matrix engine cannot help; one FMA triple per output element.
__global__ void k_gemm_l1(const float* __restrict__ x, const float* __restrict__ W,
                          float* __restrict__ h, int Nout) {
  int i = blockIdx.x * blockDim.x + threadIdx.x;
  if (i >= N_NODES * Nout) return;
  int n = i / Nout, j = i % Nout;
  const float* xp = x + (size_t)n * 3;
  h[i] = fmaf(xp[0], W[j], fmaf(xp[1], W[Nout + j], xp[2] * W[2 * Nout + j]));
}

// ------------------------------ WMMA GEMM ----------------------------------
// C[M,Nout](f32) = A[M,K](bf16, row-major) x Bt[Nout,K](bf16, row-major)^T
// Block = 256 threads = 8 waves: 2 waves in M (32 rows) x 4 waves in N.
// Each wave computes a 16x32 strip: A fragment reused across 2 WMMAs.
// Fragment layout per CDNA5 ISA (16-bit A 16x32; C/D 16x16 f32).
// Grid is sized exactly (20000%32==0, Hout%128==0, K%32==0): EXEC all-1s.
__global__ void __launch_bounds__(256)
k_gemm_wmma(const __bf16* __restrict__ A, const __bf16* __restrict__ Bt,
            float* __restrict__ C, int Nout, int K) {
  const int lane = threadIdx.x & 31;
  const int wid  = threadIdx.x >> 5;
  const int wm   = wid & 1;            // 2 waves along M
  const int wn   = wid >> 1;           // 4 waves along N
  const int row0 = blockIdx.x * 32 + wm * 16;
  const int col0 = blockIdx.y * 128 + wn * 32;
  const int half = lane >> 4;          // K-half selector per ISA layout
  const int lr   = lane & 15;          // row (A) / col (B) within tile
  const int koff = half * 8;

  const __bf16* pa  = A  + (size_t)(row0 + lr)      * K + koff;
  const __bf16* pb0 = Bt + (size_t)(col0 + lr)      * K + koff;
  const __bf16* pb1 = Bt + (size_t)(col0 + 16 + lr) * K + koff;

  v8f acc0 = {}; v8f acc1 = {};
  for (int kb = 0; kb < K; kb += 32) {
    __builtin_prefetch(pa + kb + 64, 0, 1);   // global_prefetch_b8 (L2 warm)
    v8bf alo  = *(const v8bf*)(pa  + kb);
    v8bf ahi  = *(const v8bf*)(pa  + kb + 16);
    v8bf b0lo = *(const v8bf*)(pb0 + kb);
    v8bf b0hi = *(const v8bf*)(pb0 + kb + 16);
    v8bf b1lo = *(const v8bf*)(pb1 + kb);
    v8bf b1hi = *(const v8bf*)(pb1 + kb + 16);
    v16bf a, b0, b1;
#pragma unroll
    for (int t = 0; t < 8; ++t) {
      a[t]  = alo[t];  a[8 + t]  = ahi[t];
      b0[t] = b0lo[t]; b0[8 + t] = b0hi[t];
      b1[t] = b1lo[t]; b1[8 + t] = b1hi[t];
    }
    acc0 = __builtin_amdgcn_wmma_f32_16x16x32_bf16(false, a, false, b0,
                                                   (short)0, acc0, false, false);
    acc1 = __builtin_amdgcn_wmma_f32_16x16x32_bf16(false, a, false, b1,
                                                   (short)0, acc1, false, false);
  }

  // D layout: VGPR r, lanes 0-15 -> M=r, lanes 16-31 -> M=8+r; N=lane%16.
  float* pc0 = C + (size_t)(row0 + half * 8) * Nout + col0 + lr;
  float* pc1 = pc0 + 16;
#pragma unroll
  for (int r = 0; r < 8; ++r) {
    pc0[(size_t)r * Nout] = acc0[r];
    pc1[(size_t)r * Nout] = acc1[r];
  }
}

// --------------------------- attention kernels -----------------------------

__global__ void k_alpha(const float* __restrict__ h, const float* __restrict__ a_s,
                        const float* __restrict__ a_d, float* __restrict__ als,
                        float* __restrict__ ald, int C) {
  int i = blockIdx.x * blockDim.x + threadIdx.x;      // over N*H
  if (i >= N_NODES * HEADS) return;
  int n = i >> 4, hh = i & 15;
  const float* hp  = h   + (size_t)n * HEADS * C + hh * C;
  const float* asp = a_s + hh * C;
  const float* adp = a_d + hh * C;
  float s = 0.f, d = 0.f;
#pragma unroll 4
  for (int c = 0; c < C; ++c) { float v = hp[c]; s += v * asp[c]; d += v * adp[c]; }
  als[i] = s; ald[i] = d;
}

__device__ __forceinline__ void edge_sd(const long long* __restrict__ ei,
                                        int eid, int& s, int& d) {
  if (eid < N_EDGES) { s = (int)ei[eid]; d = (int)ei[N_EDGES + eid]; }
  else               { s = d = eid - N_EDGES; }
}

__device__ __forceinline__ float atomicMaxF(float* addr, float val) {
  int* ai = (int*)addr;
  int old = *ai;
  while (__int_as_float(old) < val) {
    int assumed = old;
    old = atomicCAS(ai, assumed, __float_as_int(val));
    if (old == assumed) break;
  }
  return __int_as_float(old);
}

// pass A: e = leaky_relu(as[src]+ad[dst]); segment max into m[dst].
__global__ void k_edge_logit(const float* __restrict__ als, const float* __restrict__ ald,
                             const long long* __restrict__ ei,
                             float* __restrict__ e, float* __restrict__ m) {
  int i = blockIdx.x * blockDim.x + threadIdx.x;      // over E'*H
  if (i >= EH_TOT) return;
  int eid = i >> 4, hh = i & 15;
  int s, d; edge_sd(ei, eid, s, d);
  float v = als[s * HEADS + hh] + ald[d * HEADS + hh];
  v = (v > 0.f) ? v : v * NEG_SLOPE;
  e[i] = v;
  atomicMaxF(&m[d * HEADS + hh], v);
}

// pass B: ex = exp(e - m[dst]); segment sum into denom[dst]; e <- ex.
__global__ void k_edge_exp(const long long* __restrict__ ei, float* __restrict__ e,
                           const float* __restrict__ m, float* __restrict__ den) {
  int i = blockIdx.x * blockDim.x + threadIdx.x;
  if (i >= EH_TOT) return;
  int eid = i >> 4, hh = i & 15;
  int s, d; edge_sd(ei, eid, s, d);
  float ex = __expf(e[i] - m[d * HEADS + hh]);
  e[i] = ex;
  atomicAdd(&den[d * HEADS + hh], ex);
}

// pass C: out[dst] += h[src] * (ex / denom[dst]) per (edge, head).
__global__ void k_edge_agg(const long long* __restrict__ ei, const float* __restrict__ e,
                           const float* __restrict__ den, const float* __restrict__ h,
                           float* __restrict__ o, int C) {
  int i = blockIdx.x * blockDim.x + threadIdx.x;
  if (i >= EH_TOT) return;
  int eid = i >> 4, hh = i & 15;
  int s, d; edge_sd(ei, eid, s, d);
  float alpha = e[i] / (den[d * HEADS + hh] + 1e-16f);
  const float* hs = h + (size_t)s * HEADS * C + hh * C;
  float*       od = o + (size_t)d * HEADS * C + hh * C;
#pragma unroll 4
  for (int c = 0; c < C; ++c) atomicAdd(&od[c], hs[c] * alpha);
}

__global__ void k_bias_act(float* __restrict__ o, const float* __restrict__ b,
                           int Hout, int relu) {
  int i = blockIdx.x * blockDim.x + threadIdx.x;
  if (i >= N_NODES * Hout) return;
  float v = o[i] + b[i % Hout];
  if (relu) v = v > 0.f ? v : 0.f;
  o[i] = v;
}

__global__ void k_colsum(const float* __restrict__ o, float* __restrict__ out, int Hout) {
  int c = blockIdx.x * blockDim.x + threadIdx.x;
  if (c >= Hout) return;
  float s = 0.f;
  for (int n = 0; n < N_NODES; ++n) s += o[(size_t)n * Hout + c];
  out[c] = s;
}

// ------------------------------- launcher ----------------------------------

static inline size_t align_up(size_t x) { return (x + 255) & ~(size_t)255; }

extern "C" void kernel_launch(void* const* d_in, const int* in_sizes, int n_in,
                              void* d_out, int out_size, void* d_ws, size_t ws_size,
                              hipStream_t stream) {
  (void)in_sizes; (void)n_in; (void)out_size; (void)ws_size;
  const float*     x  = (const float*)d_in[0];
  const long long* ei = (const long long*)d_in[1];   // int64 edge_index [2, E]

  char* ws = (char*)d_ws;
  const size_t SZ_ACT = (size_t)N_NODES * 1024 * sizeof(float);
  size_t off = 0;
  float*  hbuf = (float*) (ws + off); off = align_up(off + SZ_ACT);
  float*  obuf = (float*) (ws + off); off = align_up(off + SZ_ACT);
  __bf16* x16  = (__bf16*)(ws + off); off = align_up(off + (size_t)N_NODES * 1024 * 2);
  __bf16* w16  = (__bf16*)(ws + off); off = align_up(off + (size_t)1024 * 1024 * 2);
  float*  als  = (float*) (ws + off); off = align_up(off + (size_t)N_NODES * HEADS * 4);
  float*  ald  = (float*) (ws + off); off = align_up(off + (size_t)N_NODES * HEADS * 4);
  float*  mbuf = (float*) (ws + off); off = align_up(off + (size_t)N_NODES * HEADS * 4);
  float*  den  = (float*) (ws + off); off = align_up(off + (size_t)N_NODES * HEADS * 4);
  float*  ebuf = (float*) (ws + off); off = align_up(off + (size_t)EH_TOT * 4);

  const int Cs[6] = {64, 64, 64, 64, 64, 48};
  const int TB = 256;
  const int gNH = (N_NODES * HEADS + TB - 1) / TB;
  const int gEH = (EH_TOT + TB - 1) / TB;

  for (int l = 0; l < 6; ++l) {
    const float* W  = (const float*)d_in[2 + l * 4 + 0];
    const float* as = (const float*)d_in[2 + l * 4 + 1];
    const float* ad = (const float*)d_in[2 + l * 4 + 2];
    const float* bb = (const float*)d_in[2 + l * 4 + 3];
    const int C = Cs[l];
    const int Hout = HEADS * C;            // 1024 or 768

    // --- dense projection h = x_l @ W_l ---
    if (l == 0) {
      int n = N_NODES * Hout;
      k_gemm_l1<<<(n + TB - 1) / TB, TB, 0, stream>>>(x, W, hbuf, Hout);
    } else {
      int nact = N_NODES * 1024;
      k_cvt_bf16<<<(nact + TB - 1) / TB, TB, 0, stream>>>(obuf, x16, nact);
      int nw = 1024 * Hout;
      k_cvt_w_t<<<(nw + TB - 1) / TB, TB, 0, stream>>>(W, w16, 1024, Hout);
      dim3 grid(N_NODES / 32, Hout / 128);   // 625 x {8,6}; exact tiling
      k_gemm_wmma<<<grid, TB, 0, stream>>>(x16, w16, hbuf, Hout, 1024);
    }

    // --- attention coefficients ---
    k_alpha<<<gNH, TB, 0, stream>>>(hbuf, as, ad, als, ald, C);
    k_fill_f32<<<gNH, TB, 0, stream>>>(mbuf, -3.402823466e38f, N_NODES * HEADS);
    k_fill_f32<<<gNH, TB, 0, stream>>>(den, 0.f, N_NODES * HEADS);
    {
      int n = N_NODES * Hout;
      k_fill_f32<<<(n + TB - 1) / TB, TB, 0, stream>>>(obuf, 0.f, n);
    }
    k_edge_logit<<<gEH, TB, 0, stream>>>(als, ald, ei, ebuf, mbuf);
    k_edge_exp  <<<gEH, TB, 0, stream>>>(ei, ebuf, mbuf, den);
    k_edge_agg  <<<gEH, TB, 0, stream>>>(ei, ebuf, den, hbuf, obuf, C);
    k_bias_act<<<(N_NODES * Hout + TB - 1) / TB, TB, 0, stream>>>(obuf, bb, Hout, l < 5);
  }

  k_colsum<<<3, TB, 0, stream>>>(obuf, (float*)d_out, 768);
}
